// BinaryMLP_71141838291236
// MI455X (gfx1250) — matrix operand order repaired and checked by reading, not compiled, not run on hardware
//
#include <hip/hip_runtime.h>
#include <math.h>

// ---------------------------------------------------------------------------
// Types for CDNA5 WMMA (wave32, V_WMMA_F32_16X16X32_BF16)
// ---------------------------------------------------------------------------
typedef __bf16 bf16_t;
typedef bf16_t v16bf __attribute__((ext_vector_type(16)));
typedef float  v8f   __attribute__((ext_vector_type(8)));
typedef unsigned int v4u __attribute__((ext_vector_type(4)));

union ABFrag { v16bf v; v4u q[2]; };

// Problem sizes (match reference)
#define BATCH 4096
#define D_IN  4096
#define H1    4096
#define H2    4096
#define H3    2048
#define NCLS  1000
#define NCLSP 1024   // padded to multiple of 128 for the GEMM tiler
#define BN_EPS 1e-5f
#define STAT_CHUNKS 16

// LDS row stride (elements). 40 bf16 = 80B = 20 banks: the 16 row-lanes of a
// b128 fragment read land on disjoint 4-bank groups -> conflict-free.
#define LDSS 40

// ---------------------------------------------------------------------------
// CDNA5 async global->LDS copy (GLOBAL_LOAD_ASYNC_TO_LDS_B128, ASYNCcnt).
// INST_OFFSET is added to BOTH the global and the LDS address (ISA 10.x /
// CDNA5 ISA), so one VGPR address pair serves several 16B chunks.
// ---------------------------------------------------------------------------
template <int OFF>
__device__ __forceinline__ void async_b128(unsigned lds_addr, const bf16_t* gp) {
  asm volatile("global_load_async_to_lds_b128 %0, %1, off offset:%2"
               :: "v"(lds_addr), "v"(gp), "i"(OFF) : "memory");
}
__device__ __forceinline__ void wait_async0() {
  asm volatile("s_wait_asynccnt 0x0" ::: "memory");
}

// ---------------------------------------------------------------------------
// Elementwise prep kernels
// ---------------------------------------------------------------------------
__global__ void cast_bf16_kernel(const float* __restrict__ src,
                                 bf16_t* __restrict__ dst, int n) {
  int i = blockIdx.x * blockDim.x + threadIdx.x;
  if (i < n) dst[i] = (bf16_t)src[i];
}

__global__ void cast_sign_kernel(const float* __restrict__ src,
                                 bf16_t* __restrict__ dst, int n) {
  int i = blockIdx.x * blockDim.x + threadIdx.x;
  if (i < n) dst[i] = (bf16_t)((src[i] >= 0.0f) ? 1.0f : -1.0f);
}

// W3 is (1000 x 2048); pad rows to 1024 with zeros so the GEMM tiler needs no guards.
__global__ void pad_w3_kernel(const float* __restrict__ W3,
                              bf16_t* __restrict__ dst) {
  int i = blockIdx.x * blockDim.x + threadIdx.x;          // over 1024*2048
  if (i >= NCLSP * H3) return;
  int n = i / H3;
  dst[i] = (n < NCLS) ? (bf16_t)W3[i] : (bf16_t)0.0f;
}

__global__ void pad_b3_kernel(const float* __restrict__ b3,
                              float* __restrict__ dst) {
  int i = blockIdx.x * blockDim.x + threadIdx.x;
  if (i < NCLSP) dst[i] = (i < NCLS) ? b3[i] : 0.0f;
}

// ---------------------------------------------------------------------------
// GEMM: C(f32, M x N) = A(bf16, M x K, row-major) @ W(bf16, N x K)^T + bias
// Workgroup = 256 threads = 8 waves; WG tile 256x128; wave tile 64x64 (4x4
// sub-tiles of 16x16 -> 16 WMMA per k-step per wave). K-step = 32.
// A/W tiles double-buffered in LDS, filled by async global->LDS copies.
// Per k-step sync: s_wait_asynccnt 0 + one workgroup barrier.
// ---------------------------------------------------------------------------
__global__ __launch_bounds__(256)
void gemm_bf16_wmma(const bf16_t* __restrict__ A, const bf16_t* __restrict__ W,
                    const float* __restrict__ bias, float* __restrict__ Cout,
                    int N, int K) {
  // [ bufA0 | bufA1 | bufB0 | bufB1 ]  = 2*256*40 + 2*128*40 bf16 = 60 KiB
  __shared__ __align__(16) bf16_t smem[2 * 256 * LDSS + 2 * 128 * LDSS];

  const int tid   = threadIdx.x;
  const int lane  = tid & 31;
  const int wave  = tid >> 5;
  const int waveM = wave & 3;          // 0..3 -> M offset waveM*64
  const int waveN = wave >> 2;         // 0..1 -> N offset waveN*64
  const int mblock = blockIdx.y * 256;
  const int nblock = blockIdx.x * 128;

  // Staging: A tile 256 rows x 32 k (one row = 64B per thread, 4 async b128);
  //          B tile 128 rows x 32 k (half row = 32B per thread, 2 async b128).
  const bf16_t* gA = A + (size_t)(mblock + tid) * K;
  const bf16_t* gB = W + (size_t)(nblock + (tid >> 1)) * K + ((tid & 1) << 4);

  unsigned ldsA[2], ldsB[2];
  ldsA[0] = (unsigned)(uintptr_t)(smem + tid * LDSS);
  ldsA[1] = (unsigned)(uintptr_t)(smem + 256 * LDSS + tid * LDSS);
  ldsB[0] = (unsigned)(uintptr_t)(smem + 2 * 256 * LDSS +
                                  (tid >> 1) * LDSS + ((tid & 1) << 4));
  ldsB[1] = (unsigned)(uintptr_t)(smem + 2 * 256 * LDSS + 128 * LDSS +
                                  (tid >> 1) * LDSS + ((tid & 1) << 4));

  // Fragment addressing per ISA 7.12.2 (wave32, 16-bit operands):
  //  A 16x32: lanes 0-15 hold K=0..7,16..23 of row M=lane; lanes 16-31: K=8..15,24..31.
  //  B 32x16: lanes 0-15 hold K=0..7,16..23 of col N=lane; lanes 16-31: K=8..15,24..31.
  const int halfA = (lane >> 4) << 3;  // 0 or 8
  const int koffB = (lane >> 4) << 3;  // 0 or 8
  const int rlane = lane & 15;

  auto issue = [&](int k0, int b) {
    const bf16_t* pa = gA + k0;
    const bf16_t* pb = gB + k0;
    async_b128<0>(ldsA[b], pa);
    async_b128<16>(ldsA[b], pa);
    async_b128<32>(ldsA[b], pa);
    async_b128<48>(ldsA[b], pa);
    async_b128<0>(ldsB[b], pb);
    async_b128<16>(ldsB[b], pb);
    // Pull the tile after this one toward the WGP (speculative; OOB is dropped).
    __builtin_prefetch((const void*)(pa + 32), 0, 3);
    __builtin_prefetch((const void*)(pb + 32), 0, 3);
  };

  const int ksteps = K >> 5;
  issue(0, 0);

  v8f acc[4][4] = {};

  for (int ks = 0; ks < ksteps; ++ks) {
    const int buf = ks & 1;
    // Drain my async copies into `buf`; barrier proves every wave did the same
    // AND that everyone's previous-iteration fragment reads (dscnt-drained
    // before their wmma) are done before we overwrite the other buffer below.
    wait_async0();
    __syncthreads();
    if (ks + 1 < ksteps) issue((ks + 1) << 5, buf ^ 1);

    const bf16_t* pAb = smem + buf * (256 * LDSS);
    const bf16_t* pBb = smem + 2 * 256 * LDSS + buf * (128 * LDSS);

    ABFrag af[4], bfr[4];
#pragma unroll
    for (int mi = 0; mi < 4; ++mi) {
      const bf16_t* p = pAb + (waveM * 64 + mi * 16 + rlane) * LDSS;
      af[mi].q[0] = *(const v4u*)(p + halfA);        // K = halfA..halfA+7
      af[mi].q[1] = *(const v4u*)(p + 16 + halfA);   // K = 16+halfA..
    }
#pragma unroll
    for (int ni = 0; ni < 4; ++ni) {
      const bf16_t* p = pBb + (waveN * 64 + ni * 16 + rlane) * LDSS + koffB;
      bfr[ni].q[0] = ((const v4u*)p)[0];             // K = koffB..koffB+7
      bfr[ni].q[1] = ((const v4u*)p)[2];             // K = 16+koffB..+7
    }

#pragma unroll
    for (int mi = 0; mi < 4; ++mi)
#pragma unroll
      for (int ni = 0; ni < 4; ++ni)
        acc[mi][ni] = __builtin_amdgcn_wmma_f32_16x16x32_bf16(
            false, af[mi].v, false, bfr[ni].v,
            (short)0, acc[mi][ni], false, false);
  }

  // C/D layout: VGPR v, lane l -> M = v + (l>>4)*8, N = l&15.
#pragma unroll
  for (int ni = 0; ni < 4; ++ni) {
    const int ncol = nblock + waveN * 64 + ni * 16 + rlane;
    const float bv = bias[ncol];
#pragma unroll
    for (int mi = 0; mi < 4; ++mi) {
      const int mrow = mblock + waveM * 64 + mi * 16 + ((lane >> 4) << 3);
#pragma unroll
      for (int v = 0; v < 8; ++v)
        Cout[(size_t)(mrow + v) * N + ncol] = acc[mi][ni][v] + bv;
    }
  }
}

// ---------------------------------------------------------------------------
// Per-column batch statistics in two stages (biased variance, matches
// reference h.var()). Stage 1: 16 row-chunks of partial sums. Stage 2: reduce.
// ---------------------------------------------------------------------------
__global__ void colstats_partial(const float* __restrict__ h, int M, int N,
                                 float* __restrict__ ps, float* __restrict__ ps2) {
  int c = blockIdx.x * blockDim.x + threadIdx.x;
  if (c >= N) return;
  const int rows = M / STAT_CHUNKS;
  const int r0 = blockIdx.y * rows;
  float s = 0.0f, s2 = 0.0f;
  for (int r = r0; r < r0 + rows; ++r) {
    float x = h[(size_t)r * N + c];
    s += x; s2 += x * x;
  }
  ps [(size_t)blockIdx.y * N + c] = s;
  ps2[(size_t)blockIdx.y * N + c] = s2;
}

__global__ void colstats_final(const float* __restrict__ ps,
                               const float* __restrict__ ps2, int M, int N,
                               float* __restrict__ mean, float* __restrict__ var) {
  int c = blockIdx.x * blockDim.x + threadIdx.x;
  if (c >= N) return;
  float s = 0.0f, s2 = 0.0f;
  for (int k = 0; k < STAT_CHUNKS; ++k) {
    s  += ps [(size_t)k * N + c];
    s2 += ps2[(size_t)k * N + c];
  }
  float m = s / (float)M;
  mean[c] = m;
  var[c]  = s2 / (float)M - m * m;
}

// ---------------------------------------------------------------------------
// BN (train-mode batch stats) + ReLU + cast to bf16 for the next GEMM.
// ---------------------------------------------------------------------------
__global__ void bnrelu_cast_kernel(const float* __restrict__ h,
                                   const float* __restrict__ mean,
                                   const float* __restrict__ var,
                                   const float* __restrict__ g,
                                   const float* __restrict__ beta,
                                   bf16_t* __restrict__ out,
                                   int N, int total) {
  int i = blockIdx.x * blockDim.x + threadIdx.x;
  if (i >= total) return;
  int c = i % N;
  float x = (h[i] - mean[c]) * rsqrtf(var[c] + BN_EPS) * g[c] + beta[c];
  out[i] = (bf16_t)fmaxf(x, 0.0f);
}

// ---------------------------------------------------------------------------
// log_softmax over the first NCLS of each NCLSP-wide padded logits row.
// One 256-thread block per row.
// ---------------------------------------------------------------------------
__global__ __launch_bounds__(256)
void log_softmax_kernel(const float* __restrict__ logits, float* __restrict__ out) {
  __shared__ float red[256];
  const int r = blockIdx.x;
  const float* xp = logits + (size_t)r * NCLSP;

  float lmax = -INFINITY;
  for (int c = threadIdx.x; c < NCLS; c += 256) lmax = fmaxf(lmax, xp[c]);
  red[threadIdx.x] = lmax; __syncthreads();
  for (int s = 128; s > 0; s >>= 1) {
    if (threadIdx.x < s) red[threadIdx.x] = fmaxf(red[threadIdx.x], red[threadIdx.x + s]);
    __syncthreads();
  }
  const float m = red[0]; __syncthreads();

  float lsum = 0.0f;
  for (int c = threadIdx.x; c < NCLS; c += 256) lsum += expf(xp[c] - m);
  red[threadIdx.x] = lsum; __syncthreads();
  for (int s = 128; s > 0; s >>= 1) {
    if (threadIdx.x < s) red[threadIdx.x] += red[threadIdx.x + s];
    __syncthreads();
  }
  const float lse = logf(red[0]) + m;

  for (int c = threadIdx.x; c < NCLS; c += 256)
    out[(size_t)r * NCLS + c] = xp[c] - lse;
}

// ---------------------------------------------------------------------------
// Host-side orchestration
// ---------------------------------------------------------------------------
extern "C" void kernel_launch(void* const* d_in, const int* in_sizes, int n_in,
                              void* d_out, int out_size, void* d_ws, size_t ws_size,
                              hipStream_t stream) {
  (void)in_sizes; (void)n_in; (void)out_size; (void)ws_size;

  const float* x   = (const float*)d_in[0];
  const float* W0  = (const float*)d_in[1];
  const float* b0  = (const float*)d_in[2];
  const float* g0  = (const float*)d_in[3];
  const float* be0 = (const float*)d_in[4];
  const float* Wb1 = (const float*)d_in[5];
  const float* bb1 = (const float*)d_in[6];
  const float* g1  = (const float*)d_in[7];
  const float* be1 = (const float*)d_in[8];
  const float* Wb2 = (const float*)d_in[9];
  const float* bb2 = (const float*)d_in[10];
  const float* g2  = (const float*)d_in[11];
  const float* be2 = (const float*)d_in[12];
  const float* W3  = (const float*)d_in[13];
  const float* b3  = (const float*)d_in[14];
  float* out = (float*)d_out;

  // Workspace layout (bytes, 256B-aligned chunks); total ~213 MiB.
  char* ws = (char*)d_ws;
  size_t off = 0;
  auto take = [&](size_t bytes) {
    size_t r = off; off += (bytes + 255) & ~(size_t)255; return r;
  };
  bf16_t* xbf   = (bf16_t*)(ws + take((size_t)BATCH * D_IN * 2));
  bf16_t* w0bf  = (bf16_t*)(ws + take((size_t)H1 * D_IN * 2));
  bf16_t* w1bf  = (bf16_t*)(ws + take((size_t)H2 * H1 * 2));
  bf16_t* w2bf  = (bf16_t*)(ws + take((size_t)H3 * H2 * 2));
  bf16_t* w3bf  = (bf16_t*)(ws + take((size_t)NCLSP * H3 * 2));
  float*  b3p   = (float*) (ws + take((size_t)NCLSP * 4));
  float*  meanb = (float*) (ws + take((size_t)H1 * 4));
  float*  varb  = (float*) (ws + take((size_t)H1 * 4));
  float*  psum  = (float*) (ws + take((size_t)STAT_CHUNKS * H1 * 4));
  float*  psum2 = (float*) (ws + take((size_t)STAT_CHUNKS * H1 * 4));
  float*  hbuf  = (float*) (ws + take((size_t)BATCH * H1 * 4));   // reused; holds logits last
  bf16_t* actbf = (bf16_t*)(ws + take((size_t)BATCH * H1 * 2));

  const int TPB = 256;
  auto blocks = [](long n, int t) { return (int)((n + t - 1) / t); };

  auto run_stats = [&](float* h, int N) {
    colstats_partial<<<dim3(blocks(N, TPB), STAT_CHUNKS), TPB, 0, stream>>>(
        h, BATCH, N, psum, psum2);
    colstats_final<<<blocks(N, TPB), TPB, 0, stream>>>(psum, psum2, BATCH, N,
                                                       meanb, varb);
  };

  // --- Precision prep ------------------------------------------------------
  cast_bf16_kernel<<<blocks((long)BATCH * D_IN, TPB), TPB, 0, stream>>>(x, xbf, BATCH * D_IN);
  cast_bf16_kernel<<<blocks((long)H1 * D_IN, TPB), TPB, 0, stream>>>(W0, w0bf, H1 * D_IN);
  cast_sign_kernel<<<blocks((long)H2 * H1, TPB), TPB, 0, stream>>>(Wb1, w1bf, H2 * H1);
  cast_sign_kernel<<<blocks((long)H3 * H2, TPB), TPB, 0, stream>>>(Wb2, w2bf, H3 * H2);
  pad_w3_kernel<<<blocks((long)NCLSP * H3, TPB), TPB, 0, stream>>>(W3, w3bf);
  pad_b3_kernel<<<blocks(NCLSP, TPB), TPB, 0, stream>>>(b3, b3p);

  // --- Layer 0: Linear + BN + ReLU ----------------------------------------
  gemm_bf16_wmma<<<dim3(H1 / 128, BATCH / 256), TPB, 0, stream>>>(xbf, w0bf, b0, hbuf, H1, D_IN);
  run_stats(hbuf, H1);
  bnrelu_cast_kernel<<<blocks((long)BATCH * H1, TPB), TPB, 0, stream>>>(
      hbuf, meanb, varb, g0, be0, actbf, H1, BATCH * H1);

  // --- Layer 1: BinLinear + BN + ReLU -------------------------------------
  gemm_bf16_wmma<<<dim3(H2 / 128, BATCH / 256), TPB, 0, stream>>>(actbf, w1bf, bb1, hbuf, H2, H1);
  run_stats(hbuf, H2);
  bnrelu_cast_kernel<<<blocks((long)BATCH * H2, TPB), TPB, 0, stream>>>(
      hbuf, meanb, varb, g1, be1, actbf, H2, BATCH * H2);

  // --- Layer 2: BinLinear + BN + ReLU -------------------------------------
  gemm_bf16_wmma<<<dim3(H3 / 128, BATCH / 256), TPB, 0, stream>>>(actbf, w2bf, bb2, hbuf, H3, H2);
  run_stats(hbuf, H3);
  bnrelu_cast_kernel<<<blocks((long)BATCH * H3, TPB), TPB, 0, stream>>>(
      hbuf, meanb, varb, g2, be2, actbf, H3, BATCH * H3);

  // --- Layer 3: Linear (padded N) + log_softmax ----------------------------
  gemm_bf16_wmma<<<dim3(NCLSP / 128, BATCH / 256), TPB, 0, stream>>>(actbf, w3bf, b3p, hbuf, NCLSP, H3);
  log_softmax_kernel<<<BATCH, TPB, 0, stream>>>(hbuf, out);
}
